// MoeBlock_8400956031336
// MI455X (gfx1250) — compile-verified, matrix operand order checked by reading, not run on hardware
//
#include <hip/hip_runtime.h>
#include <cstdint>
#include <cstddef>

typedef __attribute__((ext_vector_type(16))) __bf16 v16bf;
typedef __attribute__((ext_vector_type(8)))  float  v8f;

namespace {

constexpr int NT = 4096;     // B*S tokens
constexpr int D  = 1024;
constexpr int F  = 2048;
constexpr int E  = 8;
constexpr int T  = NT * 2;   // token-copies (top-2)
constexpr int TP = 9216;     // padded row space: T + E*127 rounded up

constexpr int BM = 128, BN = 64, BK = 32;

__device__ __forceinline__ unsigned short f2bf(float f) {
  unsigned int u = __float_as_uint(f);
  u += 0x7FFFu + ((u >> 16) & 1u);   // round-to-nearest-even
  return (unsigned short)(u >> 16);
}

// LDS fragment-order address for A value (m, k) within a 16x32 bf16 subtile:
// lane = (m&15) + 16*h with h = (k>>3)&1 ; element i = (k&7) + 8*((k>>4)&1)
__device__ __forceinline__ int fragA(int m, int k) {
  int h = (k >> 3) & 1;
  int i = (k & 7) + (((k >> 4) & 1) << 3);
  return ((m >> 4) << 9) + (((m & 15) + (h << 4)) << 4) + i;
}
// B value (k, n): same packing with n playing the lane role
__device__ __forceinline__ int fragB(int k, int n) {
  int h = (k >> 3) & 1;
  int i = (k & 7) + (((k >> 4) & 1) << 3);
  return ((n >> 4) << 9) + (((n & 15) + (h << 4)) << 4) + i;
}

__global__ void moe_init(int* __restrict__ counts, int* __restrict__ cursors) {
  int i = threadIdx.x;
  if (i < E) { counts[i] = 0; cursors[i] = 0; }
}

__global__ __launch_bounds__(256) void moe_gating(
    const float* __restrict__ x, const float* __restrict__ wg,
    int* __restrict__ topi, float* __restrict__ probs, int* __restrict__ counts) {
  __shared__ float swg[D * E];
  int tid = threadIdx.x;
  for (int i = tid; i < D * E; i += 256) swg[i] = wg[i];
  __syncthreads();

  int t = blockIdx.x * 256 + tid;
  const float* xr = x + (size_t)t * D;
  float acc[E] = {};
  for (int d = 0; d < D; ++d) {
    float xv = xr[d];
#pragma unroll
    for (int e = 0; e < E; ++e) acc[e] = fmaf(xv, swg[d * E + e], acc[e]);
  }
  // top-2 (ties -> lowest index, matching jax.lax.top_k)
  int e0 = 0; float l0 = acc[0];
#pragma unroll
  for (int e = 1; e < E; ++e) if (acc[e] > l0) { l0 = acc[e]; e0 = e; }
  int e1 = -1; float l1 = -3.402823466e+38f;
#pragma unroll
  for (int e = 0; e < E; ++e) if (e != e0 && acc[e] > l1) { l1 = acc[e]; e1 = e; }
  float ex = __expf(l1 - l0);
  float inv = 1.0f / (1.0f + ex);
  topi[2 * t]     = e0;  topi[2 * t + 1] = e1;
  probs[2 * t]    = inv; probs[2 * t + 1] = ex * inv;
  atomicAdd(&counts[e0], 1);
  atomicAdd(&counts[e1], 1);
}

// Prefix sum with offsets padded to BM so every expert group is 128-row aligned
__global__ void moe_scan(const int* __restrict__ counts, int* __restrict__ offsets) {
  if (threadIdx.x == 0) {
    int s = 0;
    for (int e = 0; e < E; ++e) { offsets[e] = s; s += (counts[e] + (BM - 1)) & ~(BM - 1); }
  }
}

__global__ __launch_bounds__(256) void moe_scatter(
    const int* __restrict__ topi, const int* __restrict__ offsets,
    int* __restrict__ cursors, int* __restrict__ rowtok, int* __restrict__ inv) {
  int c = blockIdx.x * 256 + threadIdx.x;
  int e = topi[c];
  int p = atomicAdd(&cursors[e], 1);
  int g = offsets[e] + p;
  rowtok[g] = c >> 1;   // source token
  inv[c] = g;           // copy -> padded sorted slot
}

// Grouped dual GEMM: h0 = Xs*W0[e], h1 = Xs*W1[e]; inter = silu(h0)*h1, written
// as bf16 in WMMA-fragment-swizzled order: inter[(slot/16)*(F/32)+ (n/32)][512]
__global__ __launch_bounds__(256) void moe_ffn1(
    const float* __restrict__ x, const float* __restrict__ w0, const float* __restrict__ w1,
    const int* __restrict__ rowtok, const int* __restrict__ counts,
    const int* __restrict__ offsets, unsigned short* __restrict__ inter) {
  int e   = blockIdx.z;
  int cnt = counts[e];
  int m0  = blockIdx.y * BM;
  if (m0 >= cnt) return;
  int n0   = blockIdx.x * BN;
  int offe = offsets[e];
  int tid = threadIdx.x, lane = tid & 31, w = tid >> 5;
  int wm = w & 3, wn = w >> 2;   // 4x2 wave grid, each wave 32x32

  __shared__ __align__(32) unsigned short sA[BM * BK];
  __shared__ __align__(32) unsigned short sB0[BN * BK];
  __shared__ __align__(32) unsigned short sB1[BN * BK];

  v8f z = {0.f, 0.f, 0.f, 0.f, 0.f, 0.f, 0.f, 0.f};
  v8f acc0[2][2], acc1[2][2];
#pragma unroll
  for (int i = 0; i < 2; ++i)
#pragma unroll
    for (int j = 0; j < 2; ++j) { acc0[i][j] = z; acc1[i][j] = z; }

  const size_t wbase = (size_t)e * D * F + n0;

  // Hoist the token gather out of the K loop: resolve the 4 row base pointers once.
  const float* aptr[4];
#pragma unroll
  for (int p = 0; p < 4; ++p) {
    int m = (tid >> 3) + p * 32;
    int row = m0 + m;
    aptr[p] = (row < cnt) ? (x + (size_t)rowtok[offe + row] * D + (tid & 7) * 4) : nullptr;
  }

  for (int k0 = 0; k0 < D; k0 += BK) {
    __syncthreads();
    {   // stage A (gathered token rows, fp32 -> bf16, fragment order)
      int kk = (tid & 7) * 4;
#pragma unroll
      for (int p = 0; p < 4; ++p) {
        int m = (tid >> 3) + p * 32;
        float4 v = make_float4(0.f, 0.f, 0.f, 0.f);
        if (aptr[p]) v = *(const float4*)(aptr[p] + k0);
        sA[fragA(m, kk + 0)] = f2bf(v.x);
        sA[fragA(m, kk + 1)] = f2bf(v.y);
        sA[fragA(m, kk + 2)] = f2bf(v.z);
        sA[fragA(m, kk + 3)] = f2bf(v.w);
      }
    }
    {   // stage B0/B1 tiles (fp32 -> bf16, fragment order)
      int n4 = (tid & 15) * 4;
#pragma unroll
      for (int p = 0; p < 2; ++p) {
        int k = (tid >> 4) + p * 16;
        const float* g0p = w0 + wbase + (size_t)(k0 + k) * F + n4;
        const float* g1p = w1 + wbase + (size_t)(k0 + k) * F + n4;
        const float4 v0 = *(const float4*)g0p;
        const float4 v1 = *(const float4*)g1p;
        __builtin_prefetch(g0p + (size_t)BK * F, 0, 0);  // next K tile (speculative)
        __builtin_prefetch(g1p + (size_t)BK * F, 0, 0);
        sB0[fragB(k, n4 + 0)] = f2bf(v0.x);
        sB0[fragB(k, n4 + 1)] = f2bf(v0.y);
        sB0[fragB(k, n4 + 2)] = f2bf(v0.z);
        sB0[fragB(k, n4 + 3)] = f2bf(v0.w);
        sB1[fragB(k, n4 + 0)] = f2bf(v1.x);
        sB1[fragB(k, n4 + 1)] = f2bf(v1.y);
        sB1[fragB(k, n4 + 2)] = f2bf(v1.z);
        sB1[fragB(k, n4 + 3)] = f2bf(v1.w);
      }
    }
    __syncthreads();

    const v16bf* pA  = (const v16bf*)sA;
    const v16bf* pB0 = (const v16bf*)sB0;
    const v16bf* pB1 = (const v16bf*)sB1;
    v16bf a[2], b0[2], b1[2];
#pragma unroll
    for (int s = 0; s < 2; ++s) a[s] = pA[(wm * 2 + s) * 32 + lane];
#pragma unroll
    for (int t2 = 0; t2 < 2; ++t2) {
      b0[t2] = pB0[(wn * 2 + t2) * 32 + lane];
      b1[t2] = pB1[(wn * 2 + t2) * 32 + lane];
    }
#pragma unroll
    for (int s = 0; s < 2; ++s)
#pragma unroll
      for (int t2 = 0; t2 < 2; ++t2) {
        acc0[s][t2] = __builtin_amdgcn_wmma_f32_16x16x32_bf16(
            false, a[s], false, b0[t2], (short)0, acc0[s][t2], false, false);
        acc1[s][t2] = __builtin_amdgcn_wmma_f32_16x16x32_bf16(
            false, a[s], false, b1[t2], (short)0, acc1[s][t2], false, false);
      }
  }

  // epilogue: inter = silu(h0)*h1, bf16, fragment-swizzled global layout
  int lh = lane >> 4, lc = lane & 15;
#pragma unroll
  for (int s = 0; s < 2; ++s)
#pragma unroll
    for (int t2 = 0; t2 < 2; ++t2)
#pragma unroll
      for (int r = 0; r < 8; ++r) {
        int m = wm * 32 + s * 16 + lh * 8 + r;
        int row = m0 + m;
        if (row < cnt) {
          int n = n0 + wn * 32 + t2 * 16 + lc;
          float h0 = acc0[s][t2][r], h1 = acc1[s][t2][r];
          float val = (h0 / (1.0f + __expf(-h0))) * h1;
          int slot = offe + row;   // 16-aligned grouping guaranteed by padding
          size_t idx = (((size_t)(slot >> 4)) * (F / 32) + (size_t)(n >> 5)) * 512
                     + (size_t)((((slot & 15) + (((n >> 3) & 1) << 4)) << 4)
                                + (n & 7) + (((n >> 4) & 1) << 3));
          inter[idx] = f2bf(val);
        }
      }
}

// Grouped GEMM: rows = inter * Wo[e] (fp32 out).
// A tiles are copied global->LDS with async-to-LDS B128 (already fragment order).
__global__ __launch_bounds__(256) void moe_ffn2(
    const unsigned short* __restrict__ inter, const float* __restrict__ wo,
    const int* __restrict__ counts, const int* __restrict__ offsets,
    float* __restrict__ rows) {
  int e   = blockIdx.z;
  int cnt = counts[e];
  int m0  = blockIdx.y * BM;
  if (m0 >= cnt) return;
  int n0   = blockIdx.x * BN;
  int offe = offsets[e];
  int tid = threadIdx.x, lane = tid & 31, w = tid >> 5;
  int wm = w & 3, wn = w >> 2;

  __shared__ __align__(32) unsigned short sA[BM * BK];
  __shared__ __align__(32) unsigned short sB[BN * BK];

  v8f z = {0.f, 0.f, 0.f, 0.f, 0.f, 0.f, 0.f, 0.f};
  v8f acc[2][2];
#pragma unroll
  for (int i = 0; i < 2; ++i)
#pragma unroll
    for (int j = 0; j < 2; ++j) acc[i][j] = z;

  const size_t wbase = (size_t)e * F * D + n0;
  const unsigned int ldsA0 = (unsigned int)(uintptr_t)(void*)sA;
  const int slotblk0 = (offe + m0) >> 4;   // 128-aligned -> 16-row blocks align

  for (int k0 = 0; k0 < F; k0 += BK) {
    __syncthreads();
    {   // async copy A subtile (wave w copies 16x32 subtile #w): 2 x B128/lane
      size_t gidx = ((size_t)(slotblk0 + w) * (F / 32) + (size_t)(k0 >> 5)) * 512;
      const unsigned short* gp = inter + gidx + lane * 8;   // 16B per lane
      unsigned int loff = ldsA0 + (unsigned int)(w * 1024 + lane * 16);
      asm volatile("global_load_async_to_lds_b128 %0, %1, off"
                   :: "v"(loff), "v"(gp) : "memory");
      asm volatile("global_load_async_to_lds_b128 %0, %1, off"
                   :: "v"(loff + 512u), "v"(gp + 256) : "memory");
    }
    {   // stage B from wo (fp32 -> bf16, fragment order)
      int n4 = (tid & 15) * 4;
#pragma unroll
      for (int p = 0; p < 2; ++p) {
        int k = (tid >> 4) + p * 16;
        const float* gbp = wo + wbase + (size_t)(k0 + k) * D + n4;
        const float4 v = *(const float4*)gbp;
        __builtin_prefetch(gbp + (size_t)BK * D, 0, 0);  // next K tile (speculative)
        sB[fragB(k, n4 + 0)] = f2bf(v.x);
        sB[fragB(k, n4 + 1)] = f2bf(v.y);
        sB[fragB(k, n4 + 2)] = f2bf(v.z);
        sB[fragB(k, n4 + 3)] = f2bf(v.w);
      }
    }
    asm volatile("s_wait_asynccnt 0" ::: "memory");
    __syncthreads();

    const v16bf* pA = (const v16bf*)sA;
    const v16bf* pB = (const v16bf*)sB;
    v16bf a[2], b[2];
#pragma unroll
    for (int s = 0; s < 2; ++s) a[s] = pA[(wm * 2 + s) * 32 + lane];
#pragma unroll
    for (int t2 = 0; t2 < 2; ++t2) b[t2] = pB[(wn * 2 + t2) * 32 + lane];
#pragma unroll
    for (int s = 0; s < 2; ++s)
#pragma unroll
      for (int t2 = 0; t2 < 2; ++t2)
        acc[s][t2] = __builtin_amdgcn_wmma_f32_16x16x32_bf16(
            false, a[s], false, b[t2], (short)0, acc[s][t2], false, false);
  }

  int lh = lane >> 4, lc = lane & 15;
#pragma unroll
  for (int s = 0; s < 2; ++s)
#pragma unroll
    for (int t2 = 0; t2 < 2; ++t2)
#pragma unroll
      for (int r = 0; r < 8; ++r) {
        int m = wm * 32 + s * 16 + lh * 8 + r;
        int row = m0 + m;
        if (row < cnt) {
          int n = n0 + wn * 32 + t2 * 16 + lc;
          rows[(size_t)(offe + row) * D + n] = acc[s][t2][r];
        }
      }
}

// out[t] = p0*rows[inv[2t]] + p1*rows[inv[2t+1]]  (fixed order -> deterministic)
__global__ __launch_bounds__(256) void moe_combine(
    const float* __restrict__ rows, const int* __restrict__ inv,
    const float* __restrict__ probs, float* __restrict__ out) {
  int idx = blockIdx.x * 256 + threadIdx.x;  // over NT * (D/4)
  int t = idx >> 8;                          // D/4 == 256
  int c = idx & 255;
  int g0 = inv[2 * t], g1 = inv[2 * t + 1];
  float p0 = probs[2 * t], p1 = probs[2 * t + 1];
  const float4* r4 = (const float4*)rows;
  float4 a = r4[(size_t)g0 * 256 + c];
  float4 b = r4[(size_t)g1 * 256 + c];
  float4 o;
  o.x = p0 * a.x + p1 * b.x;
  o.y = p0 * a.y + p1 * b.y;
  o.z = p0 * a.z + p1 * b.z;
  o.w = p0 * a.w + p1 * b.w;
  ((float4*)out)[(size_t)t * 256 + c] = o;
}

} // namespace

extern "C" void kernel_launch(void* const* d_in, const int* in_sizes, int n_in,
                              void* d_out, int out_size, void* d_ws, size_t ws_size,
                              hipStream_t stream) {
  const float* x  = (const float*)d_in[0];
  const float* wg = (const float*)d_in[1];
  const float* w0 = (const float*)d_in[2];
  const float* w1 = (const float*)d_in[3];
  const float* wo = (const float*)d_in[4];
  float* out = (float*)d_out;

  char* ws = (char*)d_ws;
  size_t off = 0;
  unsigned short* inter = (unsigned short*)(ws + off); off += (size_t)TP * F * 2;  // 36 MB
  float* rows  = (float*)(ws + off); off += (size_t)TP * D * 4;                    // 36 MB
  int*   topi  = (int*)  (ws + off); off += (size_t)T * 4;
  float* probs = (float*)(ws + off); off += (size_t)T * 4;
  int*   rowtok= (int*)  (ws + off); off += (size_t)TP * 4;
  int*   inv   = (int*)  (ws + off); off += (size_t)T * 4;
  int*   counts= (int*)  (ws + off); off += 256;
  int*   offs  = (int*)  (ws + off); off += 256;
  int*   curs  = (int*)  (ws + off); off += 256;
  if (ws_size < off) return;

  moe_init   <<<1, 64, 0, stream>>>(counts, curs);
  moe_gating <<<NT / 256, 256, 0, stream>>>(x, wg, topi, probs, counts);
  moe_scan   <<<1, 32, 0, stream>>>(counts, offs);
  moe_scatter<<<T / 256, 256, 0, stream>>>(topi, offs, curs, rowtok, inv);
  moe_ffn1   <<<dim3(F / BN, T / BM, E), 256, 0, stream>>>(x, w0, w1, rowtok, counts, offs, inter);
  moe_ffn2   <<<dim3(D / BN, T / BM, E), 256, 0, stream>>>(inter, wo, counts, offs, rows);
  moe_combine<<<(NT * (D / 4)) / 256, 256, 0, stream>>>(rows, inv, probs, out);
}
